// PCIA_65962107732786
// MI455X (gfx1250) — matrix operand order repaired and checked by reading, not compiled, not run on hardware
//
#include <hip/hip_runtime.h>
#include <hip/hip_bf16.h>

typedef __attribute__((ext_vector_type(16))) _Float16 v16h;
typedef __attribute__((ext_vector_type(8)))  float    v8f;

#define WAY   5
#define NS    5        // WAY*SHOT (SHOT=1 -> s_mean == support)
#define BTOT  80       // WAY*(SHOT+15)
#define FD    256      // FEAT_DIM
#define NPTS  4096     // N_POINTS
#define KNN   16       // SPF_KNN
#define KS    8        // SPF_KS

// ---------------------------------------------------------------------------
// Kernel 1: g = concat(support, query); gn = g / max(||g||, 1e-12)
// ---------------------------------------------------------------------------
__global__ void k_build_g(const float* __restrict__ support,
                          const float* __restrict__ query,
                          float* __restrict__ g, float* __restrict__ gn) {
  int b = blockIdx.x, d = threadIdx.x;
  __shared__ float red[FD];
  float v = (b < NS) ? support[b * FD + d] : query[(b - NS) * FD + d];
  g[b * FD + d] = v;
  red[d] = v * v;
  __syncthreads();
  for (int s = FD / 2; s > 0; s >>= 1) {
    if (d < s) red[d] += red[d + s];
    __syncthreads();
  }
  gn[b * FD + d] = v / fmaxf(sqrtf(red[0]), 1e-12f);
}

// ---------------------------------------------------------------------------
// Kernel 2: coalesced streaming pass over p_fea.
// pn2[b,n] = sum_d p^2 ; cos[b,n] = 0.5*(gn . p_n)/max(||p_n||,eps) + 0.5
// ---------------------------------------------------------------------------
__global__ void k_cos_pn2(const float* __restrict__ p_fea,
                          const float* __restrict__ gn,
                          float* __restrict__ pn2, float* __restrict__ coss) {
  int b = blockIdx.y;
  int n = blockIdx.x * 256 + threadIdx.x;
  __shared__ float sg[FD];
  sg[threadIdx.x] = gn[b * FD + threadIdx.x];
  __syncthreads();
  const float* p = p_fea + (size_t)b * FD * NPTS + n;
  float ss = 0.f, dt = 0.f;
#pragma unroll 8
  for (int d = 0; d < FD; ++d) {
    float v = p[(size_t)d * NPTS];   // contiguous across threads (n)
    ss += v * v;
    dt += sg[d] * v;
  }
  pn2[b * NPTS + n]  = ss;
  coss[b * NPTS + n] = 0.5f * dt / fmaxf(sqrtf(ss), 1e-12f) + 0.5f;
}

// ---------------------------------------------------------------------------
// Kernel 3: top-16 per batch row (descending value, ties -> lower index),
// matching jax.lax.top_k. Destroys coss (rewritten each call by k_cos_pn2).
// ---------------------------------------------------------------------------
__global__ void k_topk(float* __restrict__ coss, int* __restrict__ cidx) {
  int b = blockIdx.x, t = threadIdx.x;
  __shared__ float sv[256];
  __shared__ int   si[256];
  float* row = coss + (size_t)b * NPTS;
  for (int r = 0; r < KNN; ++r) {
    float best = -3e38f; int bi = NPTS;
    for (int n = t; n < NPTS; n += 256) {          // ascending: '>' keeps first
      float v = row[n];
      if (v > best) { best = v; bi = n; }
    }
    sv[t] = best; si[t] = bi;
    __syncthreads();
    for (int s = 128; s > 0; s >>= 1) {
      if (t < s) {
        if (sv[t + s] > sv[t] || (sv[t + s] == sv[t] && si[t + s] < si[t])) {
          sv[t] = sv[t + s]; si[t] = si[t + s];
        }
      }
      __syncthreads();
    }
    if (t == 0) { cidx[b * KNN + r] = si[0]; row[si[0]] = -3e38f; }
    __syncthreads();
  }
}

// ---------------------------------------------------------------------------
// Kernel 4: gather center rows as f16 (WMMA A operand) + their ||c||^2.
// ---------------------------------------------------------------------------
__global__ void k_gather(const float* __restrict__ p_fea,
                         const float* __restrict__ pn2,
                         const int* __restrict__ cidx,
                         _Float16* __restrict__ cf16,
                         float* __restrict__ cn2) {
  int b = blockIdx.x, d = threadIdx.x;
  const float* p = p_fea + (size_t)b * FD * NPTS;
  for (int k = 0; k < KNN; ++k) {
    int idx = cidx[b * KNN + k];
    cf16[(size_t)(b * KNN + k) * FD + d] = (_Float16)p[(size_t)d * NPTS + idx];
  }
  if (d < KNN) cn2[b * KNN + d] = pn2[b * NPTS + cidx[b * KNN + d]];
}

// ---------------------------------------------------------------------------
// Kernel 5: batched [16,256]x[256,4096] dot via v_wmma_f32_16x16x32_f16,
// composed into d2 = ||c||^2 + ||p||^2 - 2 c.p.
// One wave per block (EXEC must be all ones for WMMA). A held in VGPRs for
// all 8 K-steps; B tiles converted f32->f16 on load.
// Wave32 layouts per CDNA5 ISA 7.12.2:
//   A (16-bit 16x32): lane<16: M=lane, K = 16h + i (order used here, same as B)
//   B (32x16):        lane<16: N=lane, K = 16h + i   (h = lane>=16; order used here, same as A)
//   C/D f32 16x16:    vgpr r:  M = r + 8h, N = lane&15
// ---------------------------------------------------------------------------
__global__ void __launch_bounds__(32)
k_d2_wmma(const float* __restrict__ p_fea,
          const _Float16* __restrict__ cf16,
          const float* __restrict__ pn2,
          const float* __restrict__ cn2,
          float* __restrict__ d2) {
  int b    = blockIdx.y;
  int lane = threadIdx.x;
  int half = lane >> 4;
  int lm   = lane & 15;

  // Load full A (16x256 f16) for this batch, all 8 K-steps, into VGPRs.
  v16h A[8];
  const _Float16* arow = cf16 + (size_t)(b * KNN + lm) * FD;
#pragma unroll
  for (int t = 0; t < 8; ++t) {
    int kb = 32 * t + half * 16;
#pragma unroll
    for (int i = 0; i < 16; ++i) {
      int k = kb + i;
      A[t][i] = arow[k];
    }
  }
  float cn[8];
#pragma unroll
  for (int r = 0; r < 8; ++r) cn[r] = cn2[b * KNN + r + half * 8];

  const float* p = p_fea + (size_t)b * FD * NPTS;
  for (int nt = 0; nt < 16; ++nt) {
    int n = blockIdx.x * 256 + nt * 16 + lm;
    float pn = pn2[b * NPTS + n];
    v8f c = {};
#pragma unroll
    for (int t = 0; t < 8; ++t) {
      int kb = 32 * t + half * 16;
      v16h bt;
#pragma unroll
      for (int i = 0; i < 16; ++i)
        bt[i] = (_Float16)p[(size_t)(kb + i) * NPTS + n];
      c = __builtin_amdgcn_wmma_f32_16x16x32_f16(
          /*neg_a=*/false, A[t], /*neg_b=*/false, bt,
          /*c_mod=*/(short)0, c, /*reuse_a=*/false, /*reuse_b=*/false);
    }
#pragma unroll
    for (int r = 0; r < 8; ++r) {
      int m = r + half * 8;
      d2[(size_t)(b * KNN + m) * NPTS + n] = cn[r] + pn - 2.0f * c[r];
    }
  }
}

// ---------------------------------------------------------------------------
// Kernel 6: 8 smallest per (b,k) row (ascending, ties -> lower index),
// matching stable argsort[:, :, :8]. Destroys d2 (rewritten each call).
// ---------------------------------------------------------------------------
__global__ void k_knn(float* __restrict__ d2, int* __restrict__ lidx) {
  int row = blockIdx.x;                    // b*KNN + k : 1280 rows
  int t = threadIdx.x;
  __shared__ float sv[256];
  __shared__ int   si[256];
  float* r = d2 + (size_t)row * NPTS;
  for (int j = 0; j < KS; ++j) {
    float best = 3e38f; int bi = NPTS;
    for (int n = t; n < NPTS; n += 256) {
      float v = r[n];
      if (v < best) { best = v; bi = n; }
    }
    sv[t] = best; si[t] = bi;
    __syncthreads();
    for (int s = 128; s > 0; s >>= 1) {
      if (t < s) {
        if (sv[t + s] < sv[t] || (sv[t + s] == sv[t] && si[t + s] < si[t])) {
          sv[t] = sv[t + s]; si[t] = si[t + s];
        }
      }
      __syncthreads();
    }
    if (t == 0) { lidx[row * KS + j] = si[0]; r[si[0]] = 3e38f; }
    __syncthreads();
  }
}

// ---------------------------------------------------------------------------
// Kernel 7: fused gather + 1x1 conv (9 channels) + max over 16 centers +
// average with s_mean/query. out rows 0..4 = s_out, 5..79 = q_out.
// ---------------------------------------------------------------------------
__global__ void k_final(const float* __restrict__ p_fea,
                        const float* __restrict__ g,
                        const int* __restrict__ lidx,
                        const float* __restrict__ w,
                        float* __restrict__ out) {
  int b = blockIdx.x, d = threadIdx.x;
  __shared__ float sw[KS + 1];
  __shared__ int   sidx[KNN * KS];
  if (d < KS + 1) sw[d] = w[d];
  if (d < KNN * KS) sidx[d] = lidx[b * KNN * KS + d];
  __syncthreads();
  const float* p = p_fea + (size_t)b * FD * NPTS + (size_t)d * NPTS;
  float gv = g[b * FD + d];
  float best = -3e38f;
  for (int k = 0; k < KNN; ++k) {
    float acc = sw[KS] * gv;                  // channel 8 = broadcast g
#pragma unroll
    for (int j = 0; j < KS; ++j)
      acc += sw[j] * p[sidx[k * KS + j]];     // channels 0..7 = neighbors
    best = fmaxf(best, acc);
  }
  out[b * FD + d] = 0.5f * (best + gv);       // (spf + s_mean|query)/2
}

// ---------------------------------------------------------------------------
extern "C" void kernel_launch(void* const* d_in, const int* in_sizes, int n_in,
                              void* d_out, int out_size, void* d_ws, size_t ws_size,
                              hipStream_t stream) {
  const float* support = (const float*)d_in[0];   // [5,256]
  const float* query   = (const float*)d_in[1];   // [75,256]
  const float* p_fea   = (const float*)d_in[2];   // [80,256,4096]
  const float* conv11w = (const float*)d_in[3];   // [9]
  float* out = (float*)d_out;                     // [80,256] = s_out ++ q_out
  (void)in_sizes; (void)n_in; (void)out_size; (void)ws_size;

  // Workspace carve-up (~24.5 MB total), 256B-aligned.
  char* ws = (char*)d_ws;
  size_t o = 0;
  auto alloc = [&](size_t bytes) -> char* {
    char* p = ws + o;
    o = (o + bytes + 255) & ~(size_t)255;
    return p;
  };
  float*    g    = (float*)   alloc((size_t)BTOT * FD * 4);
  float*    gn   = (float*)   alloc((size_t)BTOT * FD * 4);
  float*    pn2  = (float*)   alloc((size_t)BTOT * NPTS * 4);
  float*    coss = (float*)   alloc((size_t)BTOT * NPTS * 4);
  int*      cidx = (int*)     alloc((size_t)BTOT * KNN * 4);
  float*    cn2  = (float*)   alloc((size_t)BTOT * KNN * 4);
  _Float16* cf16 = (_Float16*)alloc((size_t)BTOT * KNN * FD * 2);
  float*    d2   = (float*)   alloc((size_t)BTOT * KNN * NPTS * 4);
  int*      lidx = (int*)     alloc((size_t)BTOT * KNN * KS * 4);

  k_build_g<<<BTOT, FD, 0, stream>>>(support, query, g, gn);
  k_cos_pn2<<<dim3(NPTS / 256, BTOT), 256, 0, stream>>>(p_fea, gn, pn2, coss);
  k_topk<<<BTOT, 256, 0, stream>>>(coss, cidx);
  k_gather<<<BTOT, FD, 0, stream>>>(p_fea, pn2, cidx, cf16, cn2);
  k_d2_wmma<<<dim3(NPTS / 256, BTOT), 32, 0, stream>>>(p_fea, cf16, pn2, cn2, d2);
  k_knn<<<BTOT * KNN, 256, 0, stream>>>(d2, lidx);
  k_final<<<BTOT, FD, 0, stream>>>(p_fea, g, lidx, conv11w, out);
}
